// Decoder_11184094838959
// MI455X (gfx1250) — matrix operand order repaired and checked by reading, not compiled, not run on hardware
//
#include <hip/hip_runtime.h>

// ---------------------------------------------------------------------------
// 2-layer LSTM, B=32, T=8192, F=8, U=128, 4U=512.
// Persistent single-workgroup recurrent kernel, v_wmma_f32_16x16x32_bf16.
// Gate-aligned tiling: wave w owns u-tile w; its 4 accumulators are the
// i/f/g/o gate tiles for the same (m,u) lanes -> gates + cell state are
// register-resident (no z buffer in LDS).
// ---------------------------------------------------------------------------

#define Bsz   32
#define Tsz   8192
#define Fsz   8
#define Usz   128
#define Gsz   512          // 4*U

typedef __attribute__((ext_vector_type(16))) __bf16 v16bf;
typedef __attribute__((ext_vector_type(8)))  __bf16 bf16x8;
typedef __attribute__((ext_vector_type(8)))  float  v8f;

// -------------------------------------------------------------------------
// Weight conversion + swizzle into WMMA B-fragment order.
// B matrix 32x16 bf16 (wave32): element j of v16bf, lane L:
//   n = nt*16 + (L&15);  k = kq*32 + (L>>4)*16 + j
// Swizzled: frag[(nt*4+kq)*32 + lane][j], 16 bf16 contiguous per lane ->
// two global_load_b128 per fragment. 4 matrices, each 128x512 -> 65536 bf16.
// -------------------------------------------------------------------------
__global__ __launch_bounds__(256) void convert_weights(
    const float* __restrict__ W1, const float* __restrict__ U1,
    const float* __restrict__ W2, const float* __restrict__ U2,
    __bf16* __restrict__ outw)
{
    int s = blockIdx.x * 256 + threadIdx.x;      // 0 .. 262143
    int mat  = s >> 16;
    int r    = s & 65535;
    int j    = r & 15;
    int lane = (r >> 4) & 31;
    int kq   = (r >> 9) & 3;
    int nt   = r >> 11;                          // 0..31
    int k = kq * 32 + ((lane >> 4) << 4) + j;    // 0..127
    int n = nt * 16 + (lane & 15);               // 0..511
    const float* src = (mat == 0) ? W1 : (mat == 1) ? U1 : (mat == 2) ? W2 : U2;
    outw[s] = (__bf16)src[k * Gsz + n];
}

// ------------------------- fragment loaders ------------------------------
__device__ __forceinline__ v16bf cat8(bf16x8 lo, bf16x8 hi) {
    v16bf r;
#pragma unroll
    for (int j = 0; j < 8; ++j) { r[j] = lo[j]; r[j + 8] = hi[j]; }
    return r;
}

// A matrix 16x32 bf16 from row-major [32][128] bf16 in LDS.
// lane: m = mt*16 + (lane&15); elem j: k = kq*32 + half*16 + j
__device__ __forceinline__ v16bf load_a_frag(const __bf16* base, int mt,
                                             int l15, int half, int kq) {
    const __bf16* row = base + (mt * 16 + l15) * Usz + kq * 32 + half * 16;
    bf16x8 lo = *(const bf16x8*)(row);
    bf16x8 hi = *(const bf16x8*)(row + 8);
    return cat8(lo, hi);
}

// B matrix fragment from pre-swizzled global weights (L2-resident).
__device__ __forceinline__ v16bf load_b_frag(const __bf16* mat, int nt, int kq,
                                             int lane) {
    const __bf16* p = mat + ((((nt << 2) + kq) << 5) + lane) * 16;
    bf16x8 lo = *(const bf16x8*)p;
    bf16x8 hi = *(const bf16x8*)(p + 8);
    return cat8(lo, hi);
}

__device__ __forceinline__ float sigm(float v) {
    return 1.f / (1.f + __expf(-v));
}

// One LSTM layer step for this wave's u-tile.
//   z = A1@B1 + A2@B2 + bias ; gates ; c,h update.
// cst: register cell state, one v8f per mt. hout: bf16 h for next matmuls.
// gout != nullptr -> also stream f32 h to global outputs at timestep t.
__device__ __forceinline__ void lstm_layer(
    const __bf16* __restrict__ A1, const __bf16* __restrict__ A2,
    const __bf16* __restrict__ B1, const __bf16* __restrict__ B2,
    const float* __restrict__ bias, v8f* cst,
    __bf16* __restrict__ hout, float* __restrict__ gout,
    int ut, int half, int l15, int lane, int t)
{
#pragma unroll
    for (int mt = 0; mt < 2; ++mt) {
        v16bf a1[4], a2[4];
#pragma unroll
        for (int kq = 0; kq < 4; ++kq) {
            a1[kq] = load_a_frag(A1, mt, l15, half, kq);
            a2[kq] = load_a_frag(A2, mt, l15, half, kq);
        }
        v8f acc[4];   // gate order: 0=i, 1=f, 2=g(cand), 3=o
#pragma unroll
        for (int g = 0; g < 4; ++g) {
            const int nt = g * 8 + ut;
            float bb = bias[g * Usz + ut * 16 + l15];
            v8f a = {bb, bb, bb, bb, bb, bb, bb, bb};
#pragma unroll
            for (int kq = 0; kq < 4; ++kq) {
                v16bf b = load_b_frag(B1, nt, kq, lane);
                a = __builtin_amdgcn_wmma_f32_16x16x32_bf16(
                    false, a1[kq], false, b, (short)0, a, false, false);
            }
#pragma unroll
            for (int kq = 0; kq < 4; ++kq) {
                v16bf b = load_b_frag(B2, nt, kq, lane);
                a = __builtin_amdgcn_wmma_f32_16x16x32_bf16(
                    false, a2[kq], false, b, (short)0, a, false, false);
            }
            acc[g] = a;
        }
        // ---- register-resident gates: same (m,u) across the 4 accs ----
        v8f c = cst[mt];
#pragma unroll
        for (int r = 0; r < 8; ++r) {
            float cn = sigm(acc[1][r]) * c[r] + sigm(acc[0][r]) * tanhf(acc[2][r]);
            c[r] = cn;
            float h = sigm(acc[3][r]) * tanhf(cn);
            int m = mt * 16 + half * 8 + r;
            int u = ut * 16 + l15;
            hout[m * Usz + u] = (__bf16)h;
            if (gout) gout[(size_t)m * (Tsz * Usz) + (size_t)t * Usz + u] = h;
        }
        cst[mt] = c;
    }
}

// -------------------------------------------------------------------------
// Persistent sequential kernel: one workgroup, 8 wave32s, T=8192 steps.
// -------------------------------------------------------------------------
__global__ __launch_bounds__(256) void lstm_seq(
    const float* __restrict__ x,        // [32][8192][8]
    const float* __restrict__ Wl,       // [8][128]
    const float* __restrict__ bl,       // [128]
    const float* __restrict__ b1,       // [512]
    const float* __restrict__ b2,       // [512]
    const __bf16* __restrict__ wsw,     // swizzled bf16 W1,U1,W2,U2
    float* __restrict__ out)            // outputs[32][8192][128] + h1,c1,h2,c2
{
    __shared__ __align__(16) __bf16 liftb[Bsz * Usz];      // 8 KB
    __shared__ __align__(16) __bf16 h1b[2][Bsz * Usz];     // 16 KB ping-pong
    __shared__ __align__(16) __bf16 h2b[2][Bsz * Usz];     // 16 KB ping-pong
    __shared__ float s_Wl[Fsz * Usz];
    __shared__ float s_bl[Usz];
    __shared__ float s_b1[Gsz];
    __shared__ float s_b2[Gsz];
    __shared__ float s_x[Bsz * Fsz];

    const int tid = threadIdx.x;
    for (int i = tid; i < Fsz * Usz; i += 256) s_Wl[i] = Wl[i];
    if (tid < Usz) s_bl[tid] = bl[tid];
    for (int i = tid; i < Gsz; i += 256) { s_b1[i] = b1[i]; s_b2[i] = b2[i]; }
    for (int i = tid; i < Bsz * Usz; i += 256) {
        h1b[0][i] = (__bf16)0.f; h1b[1][i] = (__bf16)0.f;
        h2b[0][i] = (__bf16)0.f; h2b[1][i] = (__bf16)0.f;
    }
    v8f c1[2], c2[2];
#pragma unroll
    for (int mt = 0; mt < 2; ++mt) {
        c1[mt] = (v8f){0.f, 0.f, 0.f, 0.f, 0.f, 0.f, 0.f, 0.f};
        c2[mt] = (v8f){0.f, 0.f, 0.f, 0.f, 0.f, 0.f, 0.f, 0.f};
    }
    __syncthreads();

    const int wave = tid >> 5;
    const int lane = tid & 31;
    const int half = lane >> 4;
    const int l15  = lane & 15;
    const int ut   = wave;              // this wave's u-tile (0..7)

    const __bf16* W1s = wsw;
    const __bf16* U1s = wsw + 65536;
    const __bf16* W2s = wsw + 2 * 65536;
    const __bf16* U2s = wsw + 3 * 65536;

#pragma clang loop unroll(disable)
    for (int t = 0; t < Tsz; ++t) {
        // Opaque memory barrier: stops LICM from hoisting the (loop-
        // invariant) weight-fragment loads out of the t-loop and spilling
        // 500+ VGPRs to scratch. Weights stay L2-resident instead.
        asm volatile("" ::: "memory");

        const int rb = t & 1, wb = rb ^ 1;

        // ---- stage 1: x_t -> LDS; lift = x_t @ W_lift + b_lift (bf16) ----
        {
            int m = tid >> 3, f = tid & 7;
            s_x[tid] = x[(size_t)m * (Tsz * Fsz) + (size_t)t * Fsz + f];
        }
        __syncthreads();
#pragma unroll 4
        for (int i = 0; i < 16; ++i) {
            int e = tid + (i << 8);
            int m = e >> 7, u = e & 127;
            float acc = s_bl[u];
#pragma unroll
            for (int f = 0; f < Fsz; ++f)
                acc += s_x[(m << 3) + f] * s_Wl[f * Usz + u];
            liftb[e] = (__bf16)acc;
        }
        __syncthreads();

        // ---- layer 1: z1 = lift@W1 + h1@U1 ; gates ; h1' ----
        lstm_layer(liftb, h1b[rb], W1s, U1s, s_b1, c1, h1b[wb],
                   (float*)nullptr, ut, half, l15, lane, t);
        __syncthreads();

        // ---- layer 2: z2 = h1'@W2 + h2@U2 ; gates ; h2' (+ output) ----
        lstm_layer(h1b[wb], h2b[rb], W2s, U2s, s_b2, c2, h2b[wb],
                   out, ut, half, l15, lane, t);
        __syncthreads();
    }

    // ---- final states: (outputs, h1, c1, h2, c2) ----
    const size_t base = (size_t)Bsz * Tsz * Usz;
    const int fin = ((Tsz - 1) & 1) ^ 1;    // last-written ping-pong buffer
    for (int i = tid; i < Bsz * Usz; i += 256) {
        out[base + i]        = (float)h1b[fin][i];
        out[base + 8192 + i] = (float)h2b[fin][i];
    }
#pragma unroll
    for (int mt = 0; mt < 2; ++mt) {
#pragma unroll
        for (int r = 0; r < 8; ++r) {
            int m = mt * 16 + half * 8 + r;
            int u = ut * 16 + l15;
            out[base + 4096 + m * Usz + u]  = c1[mt][r];
            out[base + 12288 + m * Usz + u] = c2[mt][r];
        }
    }
}

extern "C" void kernel_launch(void* const* d_in, const int* in_sizes, int n_in,
                              void* d_out, int out_size, void* d_ws, size_t ws_size,
                              hipStream_t stream) {
    (void)in_sizes; (void)n_in; (void)out_size; (void)ws_size;
    const float* x      = (const float*)d_in[0];
    const float* W_lift = (const float*)d_in[1];
    const float* b_lift = (const float*)d_in[2];
    const float* W1     = (const float*)d_in[3];
    const float* U1     = (const float*)d_in[4];
    const float* b1     = (const float*)d_in[5];
    const float* W2     = (const float*)d_in[6];
    const float* U2     = (const float*)d_in[7];
    const float* b2     = (const float*)d_in[8];

    __bf16* wsw = (__bf16*)d_ws;   // 4 * 65536 bf16 = 512 KB scratch

    convert_weights<<<1024, 256, 0, stream>>>(W1, U1, W2, U2, wsw);
    lstm_seq<<<1, 256, 0, stream>>>(x, W_lift, b_lift, b1, b2, wsw,
                                    (float*)d_out);
}
